// NetVLAD_83391085019694
// MI455X (gfx1250) — matrix operand order repaired and checked by reading, not compiled, not run on hardware
//
#include <hip/hip_runtime.h>
#include <hip/hip_bf16.h>

typedef __attribute__((ext_vector_type(16))) _Float16 v16h;
typedef __attribute__((ext_vector_type(8)))  _Float16 v8h;
typedef __attribute__((ext_vector_type(8)))  float    v8f;
typedef __attribute__((ext_vector_type(2)))  float    v2f;

#define C_DIM 512
#define K_DIM 64
#define H_DIM 192
#define W_DIM 192
#define L_DIM (H_DIM * W_DIM)      // 36864
#define NSPLIT 144
#define LSPLIT (L_DIM / NSPLIT)    // 256

// ---------------------------------------------------------------- kernel 1
// inv_norm[l] = 1 / max(||x[:,l]||, 1e-12)
__global__ void k_invnorm(const float* __restrict__ x, float* __restrict__ inv_norm) {
    int l = blockIdx.x * blockDim.x + threadIdx.x;
    if (l >= L_DIM) return;
    float s = 0.f;
    for (int c = 0; c < C_DIM; ++c) {
        float v = x[c * L_DIM + l];
        s += v * v;
    }
    inv_norm[l] = 1.0f / fmaxf(sqrtf(s), 1e-12f);
}

// ---------------------------------------------------------------- kernel 1b
__global__ void k_wtof16(const float* __restrict__ w, _Float16* __restrict__ wh) {
    int i = blockIdx.x * blockDim.x + threadIdx.x;
    if (i < K_DIM * C_DIM) wh[i] = (_Float16)w[i];
}

// A fragment (16x32 f16, M x K): lanes 0-15 / 16-31 both carry M=lane&15.
// elem e<8 -> K = 8*hi + e ; elem e>=8 -> K = 16 + 8*hi + (e-8). Contiguous
// runs of 8 halves -> two 16-byte vector loads.
__device__ inline v16h load_a_frag(const _Float16* __restrict__ wh, int row, int c0, int hi) {
    const v8h* p0 = (const v8h*)(wh + row * C_DIM + c0 + 8 * hi);
    const v8h* p1 = (const v8h*)(wh + row * C_DIM + c0 + 16 + 8 * hi);
    v8h lo = *p0, hi8 = *p1;
    v16h a;
#pragma unroll
    for (int e = 0; e < 8; ++e) { a[e] = lo[e]; a[8 + e] = hi8[e]; }
    return a;
}

// ---------------------------------------------------------------- kernel 2
// Fused: logits = Wf16 x (x * inv_norm)  via v_wmma_f32_16x16x32_f16,
// then per-pixel softmax over K=64 + top-2 through LDS.
// One wave computes a 64(k) x 16(pixel) logits panel (4 accumulators).
__global__ __launch_bounds__(128) void k_logits_softmax(
        const float* __restrict__ x, const float* __restrict__ inv_norm,
        const _Float16* __restrict__ wh,
        float* __restrict__ soft, unsigned int* __restrict__ top2) {
    __shared__ float smem[4][K_DIM][16];
    const int wave = threadIdx.x >> 5;
    const int lane = threadIdx.x & 31;
    const int n    = lane & 15;
    const int hi   = lane >> 4;
    const int l0   = (blockIdx.x * 4 + wave) * 16;
    const int l    = l0 + n;
    const float inv = inv_norm[l];

    v8f acc0 = {}, acc1 = {}, acc2 = {}, acc3 = {};
    for (int c0 = 0; c0 < C_DIM; c0 += 32) {
        // B fragment (32x16 f16, K x N): lane carries column N=lane&15,
        // elem e<8 -> K = 8*hi + e ; elem e>=8 -> K = 16 + 8*hi + (e-8). Normalize on the fly.
        v16h b;
#pragma unroll
        for (int e = 0; e < 16; ++e) {
            int c = c0 + 8 * hi + e + (e >= 8 ? 8 : 0);
            b[e] = (_Float16)(x[c * L_DIM + l] * inv);
        }
        v16h a0 = load_a_frag(wh,  0 + n, c0, hi);
        v16h a1 = load_a_frag(wh, 16 + n, c0, hi);
        v16h a2 = load_a_frag(wh, 32 + n, c0, hi);
        v16h a3 = load_a_frag(wh, 48 + n, c0, hi);
        acc0 = __builtin_amdgcn_wmma_f32_16x16x32_f16(false, a0, false, b, (short)0, acc0, false, false);
        acc1 = __builtin_amdgcn_wmma_f32_16x16x32_f16(false, a1, false, b, (short)0, acc1, false, false);
        acc2 = __builtin_amdgcn_wmma_f32_16x16x32_f16(false, a2, false, b, (short)0, acc2, false, false);
        acc3 = __builtin_amdgcn_wmma_f32_16x16x32_f16(false, a3, false, b, (short)0, acc3, false, false);
    }
    // C/D layout: VGPR r -> M = r (lanes 0-15) / r+8 (lanes 16-31), N = lane&15
#pragma unroll
    for (int r = 0; r < 8; ++r) {
        smem[wave][ 0 + r + 8 * hi][n] = acc0[r];
        smem[wave][16 + r + 8 * hi][n] = acc1[r];
        smem[wave][32 + r + 8 * hi][n] = acc2[r];
        smem[wave][48 + r + 8 * hi][n] = acc3[r];
    }
    __syncthreads();

    if (lane < 16) {
        const int lp = l0 + lane;
        float m1 = -3.0e38f, m2 = -3.0e38f;
        int   i1 = 0,        i2 = 0;
        for (int k = 0; k < K_DIM; ++k) {
            float v = smem[wave][k][lane];
            if (v > m1)      { m2 = m1; i2 = i1; m1 = v; i1 = k; }
            else if (v > m2) { m2 = v;  i2 = k; }
        }
        float se = 0.f;
        for (int k = 0; k < K_DIM; ++k) {
            float e = expf(smem[wave][k][lane] - m1);
            smem[wave][k][lane] = e;
            se += e;
        }
        float inv_se = 1.0f / se;
        for (int k = 0; k < K_DIM; ++k)
            soft[k * L_DIM + lp] = smem[wave][k][lane] * inv_se;
        top2[lp] = (unsigned)i1 | ((unsigned)i2 << 8);
    }
}

// ---------------------------------------------------------------- kernel 3
// weight[k,p] = soft[k,p] * #{q in 3x3(p) : k in top2(q)} * borderdist(p)^4
__global__ void k_weight(const float* __restrict__ soft,
                         const unsigned* __restrict__ top2,
                         float* __restrict__ weight) {
    int l = blockIdx.x * blockDim.x + threadIdx.x;
    if (l >= L_DIM) return;
    int i = l / W_DIM, j = l % W_DIM;
    unsigned nb[9];
    int t = 0;
#pragma unroll
    for (int di = -1; di <= 1; ++di) {
#pragma unroll
        for (int dj = -1; dj <= 1; ++dj) {
            int ii = i + di, jj = j + dj;
            bool ok = (ii >= 0) && (ii < H_DIM) && (jj >= 0) && (jj < W_DIM);
            nb[t] = ok ? top2[ii * W_DIM + jj] : 0xFFFFu;
            ++t;
        }
    }
    int mi = min(i, H_DIM - 1 - i), mj = min(j, W_DIM - 1 - j);
    float mm = (float)min(mi, mj);
    float m2 = mm * mm;
    float mask = m2 * m2;
    for (int k = 0; k < K_DIM; ++k) {
        int cnt = 0;
#pragma unroll
        for (int q = 0; q < 9; ++q) {
            cnt += ((nb[q] & 0xFFu) == (unsigned)k);
            cnt += ((nb[q] >> 8)    == (unsigned)k);
        }
        weight[k * L_DIM + l] = soft[k * L_DIM + l] * (float)cnt * mask;
    }
}

// ---------------------------------------------------------------- kernel 3b
__global__ void k_ksum(const float* __restrict__ weight, float* __restrict__ ksum) {
    __shared__ float red[256];
    int k = blockIdx.x;
    float s = 0.f;
    for (int l = threadIdx.x; l < L_DIM; l += 256) s += weight[k * L_DIM + l];
    red[threadIdx.x] = s;
    __syncthreads();
    for (int o = 128; o > 0; o >>= 1) {
        if (threadIdx.x < (unsigned)o) red[threadIdx.x] += red[threadIdx.x + o];
        __syncthreads();
    }
    if (threadIdx.x == 0) ksum[k] = red[0];
}

// ---------------------------------------------------------------- kernel 5
// vlad partial GEMM: D[k,c] += sum_l weight[k,l] * (x[c,l]*inv[l]) in fp32
// via v_wmma_f32_16x16x4_f32. All four K tiles are computed by one wave so
// the B fragment (x * inv_norm) is loaded once and fed to 4 WMMAs.
// L split across gridDim.y into deterministic partials (no float atomics).
__global__ __launch_bounds__(32) void k_vlad_gemm(
        const float* __restrict__ weight, const float* __restrict__ x,
        const float* __restrict__ inv_norm, float* __restrict__ partial) {
    const int lane = threadIdx.x & 31;
    const int n    = lane & 15;
    const int hi   = lane >> 4;
    const int ct = blockIdx.x;   // 0..31  (C tiles)
    const int z  = blockIdx.y;   // 0..NSPLIT-1
    const float* w0   = weight + (size_t)( 0 + n) * L_DIM;  // A: M = lane&15
    const float* w1   = weight + (size_t)(16 + n) * L_DIM;
    const float* w2   = weight + (size_t)(32 + n) * L_DIM;
    const float* w3   = weight + (size_t)(48 + n) * L_DIM;
    const float* xrow = x      + (size_t)(ct * 16 + n) * L_DIM;  // B: N = lane&15
    const int lbase = z * LSPLIT + 2 * hi;  // elem v -> K = 2*hi + v

    v8f acc0 = {}, acc1 = {}, acc2 = {}, acc3 = {};
    for (int t = 0; t < LSPLIT; t += 4) {
        int l = lbase + t;
        // prefetch one cacheline group ahead on the streaming operands
        __builtin_prefetch(xrow + l + 64, 0, 0);
        __builtin_prefetch(w0 + l + 64, 0, 0);
        v2f b;
        b.x = xrow[l]     * inv_norm[l];
        b.y = xrow[l + 1] * inv_norm[l + 1];
        v2f a0, a1, a2, a3;
        a0.x = w0[l]; a0.y = w0[l + 1];
        a1.x = w1[l]; a1.y = w1[l + 1];
        a2.x = w2[l]; a2.y = w2[l + 1];
        a3.x = w3[l]; a3.y = w3[l + 1];
        acc0 = __builtin_amdgcn_wmma_f32_16x16x4_f32(false, a0, false, b, (short)0, acc0, false, false);
        acc1 = __builtin_amdgcn_wmma_f32_16x16x4_f32(false, a1, false, b, (short)0, acc1, false, false);
        acc2 = __builtin_amdgcn_wmma_f32_16x16x4_f32(false, a2, false, b, (short)0, acc2, false, false);
        acc3 = __builtin_amdgcn_wmma_f32_16x16x4_f32(false, a3, false, b, (short)0, acc3, false, false);
    }
    float* out = partial + (size_t)z * (K_DIM * C_DIM) + ct * 16 + n;
#pragma unroll
    for (int r = 0; r < 8; ++r) {
        out[( 0 + r + 8 * hi) * C_DIM] = acc0[r];
        out[(16 + r + 8 * hi) * C_DIM] = acc1[r];
        out[(32 + r + 8 * hi) * C_DIM] = acc2[r];
        out[(48 + r + 8 * hi) * C_DIM] = acc3[r];
    }
}

// ---------------------------------------------------------------- kernel 5b
__global__ void k_vlad_combine(const float* __restrict__ partial,
                               const float* __restrict__ ksum,
                               const float* __restrict__ centroids,
                               float* __restrict__ vlad) {
    int i = blockIdx.x * blockDim.x + threadIdx.x;
    if (i >= K_DIM * C_DIM) return;
    float s = 0.f;
    for (int z = 0; z < NSPLIT; ++z) s += partial[(size_t)z * (K_DIM * C_DIM) + i];
    int k = i >> 9;  // / C_DIM
    vlad[i] = s - ksum[k] * centroids[i];
}

// ---------------------------------------------------------------- kernel 6
__global__ void k_rownorm(const float* __restrict__ vlad, float* __restrict__ out,
                          float* __restrict__ rowsq) {
    __shared__ float red[256];
    int k = blockIdx.x, t = threadIdx.x;
    float a = vlad[k * C_DIM + t];
    float b = vlad[k * C_DIM + t + 256];
    red[t] = a * a + b * b;
    __syncthreads();
    for (int o = 128; o > 0; o >>= 1) {
        if (t < o) red[t] += red[t + o];
        __syncthreads();
    }
    float sumsq = red[0];
    float inv = 1.0f / fmaxf(sqrtf(sumsq), 1e-12f);
    out[k * C_DIM + t]       = a * inv;
    out[k * C_DIM + t + 256] = b * inv;
    if (t == 0) rowsq[k] = sumsq * inv * inv;
}

// ---------------------------------------------------------------- kernel 7
__global__ void k_finalnorm(const float* __restrict__ rowsq, float* __restrict__ out) {
    __shared__ float ginv;
    int t = threadIdx.x;
    if (t == 0) {
        float s = 0.f;
        for (int k = 0; k < K_DIM; ++k) s += rowsq[k];
        ginv = 1.0f / fmaxf(sqrtf(s), 1e-12f);
    }
    __syncthreads();
    float g = ginv;
    for (int i = t; i < K_DIM * C_DIM; i += blockDim.x) out[i] *= g;
}

// ----------------------------------------------------------------
extern "C" void kernel_launch(void* const* d_in, const int* in_sizes, int n_in,
                              void* d_out, int out_size, void* d_ws, size_t ws_size,
                              hipStream_t stream) {
    const float* x         = (const float*)d_in[0];   // (512,192,192)
    const float* conv_w    = (const float*)d_in[1];   // (64,512)
    const float* centroids = (const float*)d_in[2];   // (64,512)
    float* out = (float*)d_out;                       // 64*512 floats

    char* ws = (char*)d_ws;
    size_t off = 0;
    auto alloc = [&](size_t bytes) -> void* {
        void* p = ws + off;
        off += (bytes + 255) & ~(size_t)255;
        return p;
    };
    float*     inv_norm = (float*)alloc((size_t)L_DIM * 4);
    _Float16*  wh       = (_Float16*)alloc((size_t)K_DIM * C_DIM * 2);
    float*     soft     = (float*)alloc((size_t)K_DIM * L_DIM * 4);
    unsigned*  top2     = (unsigned*)alloc((size_t)L_DIM * 4);
    float*     weight   = (float*)alloc((size_t)K_DIM * L_DIM * 4);
    float*     ksum     = (float*)alloc(64 * 4);
    float*     rowsq    = (float*)alloc(64 * 4);
    float*     vlad     = (float*)alloc((size_t)K_DIM * C_DIM * 4);
    float*     partial  = (float*)alloc((size_t)NSPLIT * K_DIM * C_DIM * 4);
    (void)ws_size; (void)in_sizes; (void)n_in; (void)out_size;

    k_invnorm<<<(L_DIM + 255) / 256, 256, 0, stream>>>(x, inv_norm);
    k_wtof16<<<(K_DIM * C_DIM + 255) / 256, 256, 0, stream>>>(conv_w, wh);
    k_logits_softmax<<<L_DIM / 64, 128, 0, stream>>>(x, inv_norm, wh, soft, top2);
    k_weight<<<(L_DIM + 255) / 256, 256, 0, stream>>>(soft, top2, weight);
    k_ksum<<<K_DIM, 256, 0, stream>>>(weight, ksum);
    dim3 g2(C_DIM / 16, NSPLIT);
    k_vlad_gemm<<<g2, 32, 0, stream>>>(weight, x, inv_norm, partial);
    k_vlad_combine<<<(K_DIM * C_DIM + 255) / 256, 256, 0, stream>>>(partial, ksum, centroids, vlad);
    k_rownorm<<<K_DIM, 256, 0, stream>>>(vlad, out, rowsq);
    k_finalnorm<<<1, 256, 0, stream>>>(rowsq, out);
}
